// Net_15152644620734
// MI455X (gfx1250) — compile-verified, matrix operand order checked by reading, not run on hardware
//
#include <hip/hip_runtime.h>

// ---------------------------------------------------------------------------
// Fused sparse conv (3x3, 1->64, masked) + sparse inverse conv (3x3, 64->32,
// masked to original active set), NCHW output.
//   stage 1: conv1 as GEMM  D[ch][pix] = W1^T x xtaps  on v_wmma_f32_16x16x4_f32
//   stage 2: transpose-conv as GEMM (K = 9*64 = 576)   on v_wmma_f32_16x16x32_bf16
// W2 is pre-converted once (prep kernel) to bf16 in the WMMA-transposed
// layout; blocks stage it into LDS with GLOBAL_LOAD_ASYNC_TO_LDS_B128,
// overlapped with the stage-1 matrix work.
// ---------------------------------------------------------------------------

typedef __bf16 v16bf __attribute__((ext_vector_type(16)));
typedef float  v8f   __attribute__((ext_vector_type(8)));
typedef float  v4f   __attribute__((ext_vector_type(4)));
typedef float  v2f   __attribute__((ext_vector_type(2)));
typedef int    v4i   __attribute__((ext_vector_type(4)));
typedef unsigned int v4u __attribute__((ext_vector_type(4)));

__device__ __forceinline__ unsigned short f2bf(float f) {
    union { float f; unsigned int u; } c; c.f = f;
    unsigned int r = c.u + 0x7FFFu + ((c.u >> 16) & 1u);   // round-to-nearest-even
    return (unsigned short)(r >> 16);
}

// Pack two f32 -> packed bf16x2.  Preferred: v_cvt_pk_bf16_f32 (1 op, RNE).
// Fallback: a single v_perm_b32 packing the two high halves (truncation).
__device__ __forceinline__ unsigned int pack_bf(float a, float b) {
#if __has_builtin(__builtin_amdgcn_cvt_pk_bf16_f32)
    typedef __bf16 v2bf __attribute__((ext_vector_type(2)));
    union { v2bf v; unsigned int u; } c;
    c.v = __builtin_amdgcn_cvt_pk_bf16_f32(a, b);
    return c.u;
#else
    union { float f; unsigned int u; } ca, cb; ca.f = a; cb.f = b;
    // dst = { a[31:16], b[31:16] } : src1 bytes 0-3 = a, src0 bytes 4-7 = b
    return __builtin_amdgcn_perm(cb.u, ca.u, 0x07060302u);
#endif
}

// 16-byte global -> LDS copy; async (ASYNCcnt-tracked, no VGPR round trip).
#if __has_builtin(__builtin_amdgcn_global_load_async_to_lds_b128)
#define ASYNC_LDS 1
#else
#define ASYNC_LDS 0
#endif

__device__ __forceinline__ void copy16_g2l(const void* g, void* l) {
#if ASYNC_LDS
    __builtin_amdgcn_global_load_async_to_lds_b128(
        (__attribute__((address_space(1))) v4i*)g,
        (__attribute__((address_space(3))) v4i*)l, 0, 0);
#else
    *(v4u*)l = *(const v4u*)g;
#endif
}

__device__ __forceinline__ void wait_g2l() {
#if ASYNC_LDS
#if __has_builtin(__builtin_amdgcn_s_wait_asynccnt)
    __builtin_amdgcn_s_wait_asynccnt(0);
#else
    asm volatile("s_wait_asynccnt 0" ::: "memory");
#endif
#endif
}

// 16x32 bf16 A/B fragment from an LDS row with 32 contiguous K values.
// ISA 16-bit layout: lanes 0-15 hold K 0-7 / 16-23; lanes 16-31 K 8-15 / 24-31.
__device__ __forceinline__ v16bf load_frag(const unsigned short* base, int kh) {
    union { v16bf h; v4f q[2]; } f;
    f.q[0] = *(const v4f*)(base + (kh << 3));
    f.q[1] = *(const v4f*)(base + 16 + (kh << 3));
    return f.h;
}

#define WMMA_BF16(A, B, C) \
    __builtin_amdgcn_wmma_f32_16x16x32_bf16(false, (A), false, (B), (short)0, (C), false, false)
#define WMMA_F32K4(A, B, C) \
    __builtin_amdgcn_wmma_f32_16x16x4_f32(false, (A), false, (B), (short)0, (C), false, false)

// ---- one-shot: W2 [3,3,64,32] f32 -> bf16, layout [tap][cb][co][k] ---------
__global__ __launch_bounds__(256)
void prep_w2_bf16(const float* __restrict__ w2g, unsigned short* __restrict__ w2b) {
    int t = blockIdx.x * 256 + threadIdx.x;      // 9*2*32*32 = 18432 elements
    if (t >= 18432) return;
    int k   = t & 31;
    int co  = (t >> 5) & 31;
    int cbb = (t >> 10) & 1;
    int tap = t >> 11;
    float v = w2g[((size_t)tap * 64 + cbb * 32 + k) * 32 + co];
    w2b[t] = f2bf(v);
}

__global__ __launch_bounds__(256)
void sparse_conv_fused(const float* __restrict__ xg,            // [B,64,64,1]
                       const float* __restrict__ w1g,           // [3,3,1,64]
                       const float* __restrict__ b1g,           // [64]
                       const float* __restrict__ w2g,           // [3,3,64,32]
                       const float* __restrict__ b2g,           // [32]
                       const unsigned short* __restrict__ w2b,  // bf16 prepped (or null)
                       float* __restrict__ outg) {              // [B,32,64,64] NCHW
    __shared__ __align__(16) float          xs[20][20];      // x halo patch    1600B
    __shared__ __align__(16) float          w1pair[3][64][4];// W1^T k-pairs    3072B
    __shared__ __align__(16) float          b1s[64];         //                  256B
    __shared__ __align__(16) float          b2s[32];         //                  128B
    __shared__ __align__(16) float          a_lds[336][12];  // per-pixel taps 16128B
    __shared__ __align__(16) float          keep_lds[336];   // m1*valid        1344B
    __shared__ __align__(16) unsigned short ys[336][32];     // bf16 y slice   21504B
    __shared__ __align__(16) unsigned short w2s[9][32][32];  // bf16 W2^T      18432B

    const int tid  = threadIdx.x;
    const int Q0   = blockIdx.x * 16;   // patch col origin
    const int P0   = blockIdx.y * 16;   // patch row origin
    const int bimg = blockIdx.z;

    // ---- phase 0: async-stage b1/b2; build W1^T pairs; stage x halo --------
    if (tid < 16)       copy16_g2l(b1g + tid * 4,        &b1s[0] + tid * 4);
    else if (tid < 24)  copy16_g2l(b2g + (tid - 16) * 4, &b2s[0] + (tid - 16) * 4);

    for (int t = tid; t < 768; t += 256) {       // w1pair[ks][c][kk], zero pad k>=9
        int kk = t & 3, c = (t >> 2) & 63, ks = t >> 8;
        int k = ks * 4 + kk;
        w1pair[ks][c][kk] = (k < 9) ? w1g[k * 64 + c] : 0.0f;
    }
    for (int t = tid; t < 400; t += 256) {
        int s = t / 20, w = t - s * 20;
        int row = P0 + s - 2, col = Q0 + w - 2;
        float v = 0.0f;
        if (row >= 0 && row < 64 && col >= 0 && col < 64)
            v = xg[((size_t)bimg * 64 + row) * 64 + col];
        xs[s][w] = v;
    }
    wait_g2l();
    __syncthreads();

    // ---- phase 1: per-pixel tap vectors (stage-1 B matrix) + keep mask -----
    for (int p = tid; p < 324; p += 256) {
        int u = p / 18, v = p - u * 18;
        int U = P0 + u - 2, V = Q0 + v - 2;
        bool valid = (U >= 0) && (U < 62) && (V >= 0) && (V < 62);
        float xv[9];
        bool  any = false;
        #pragma unroll
        for (int a = 0; a < 3; ++a)
            #pragma unroll
            for (int c = 0; c < 3; ++c) {
                float t = xs[u + a][v + c];
                xv[a * 3 + c] = t;
                any = any || (t != 0.0f);
            }
        v4f q0 = {xv[0], xv[1], xv[2], xv[3]};
        v4f q1 = {xv[4], xv[5], xv[6], xv[7]};
        v4f q2 = {xv[8], 0.0f, 0.0f, 0.0f};
        *(v4f*)&a_lds[p][0] = q0;
        *(v4f*)&a_lds[p][4] = q1;
        *(v4f*)&a_lds[p][8] = q2;
        keep_lds[p] = (valid && any) ? 1.0f : 0.0f;
    }
    if (tid < 12) keep_lds[324 + tid] = 0.0f;    // pad tiles
    __syncthreads();

    const int wv   = tid >> 5;     // wave id 0..7
    const int lane = tid & 31;
    const int mrow = lane & 15;    // row (A) / col (B/D) index inside a tile
    const int kh   = lane >> 4;    // K-half / M-half selector

    v8f acc00 = {0,0,0,0,0,0,0,0}; // stage-2: row wv*2,   co 0..15
    v8f acc01 = {0,0,0,0,0,0,0,0}; //          row wv*2,   co 16..31
    v8f acc10 = {0,0,0,0,0,0,0,0}; //          row wv*2+1, co 0..15
    v8f acc11 = {0,0,0,0,0,0,0,0}; //          row wv*2+1, co 16..31

    // ---- K split into two 32-channel slices --------------------------------
    for (int cb = 0; cb < 2; ++cb) {
        // (a) issue async W2-slice copy; it overlaps the stage-1 matrix work.
        if (w2b != nullptr) {
            for (int t = tid; t < 1152; t += 256) {        // 9 taps * 128 x 16B
                int tap = t >> 7, rem = t & 127;
                copy16_g2l(w2b + (((size_t)(tap * 2 + cb)) << 10) + (rem << 3),
                           &w2s[0][0][0] + ((size_t)tap << 10) + (rem << 3));
            }
        } else {
            for (int t = tid; t < 9216; t += 256) {        // fallback: convert
                int co = t & 31, k = (t >> 5) & 31, tap = t >> 10;
                w2s[tap][co][k] =
                    f2bf(w2g[((size_t)tap * 64 + cb * 32 + k) * 32 + co]);
            }
        }

        // (b) stage-1 WMMA: D[ch16][pix16] = W1^T x xtaps, then (+b1)*keep,
        //     bf16-packed straight into the stage-2 A layout ys[pix][ch].
        {
            // A fragments (W1^T rows = channels) for both ch-tiles, 3 k-steps.
            v2f afr[2][3];
            #pragma unroll
            for (int ct = 0; ct < 2; ++ct)
                #pragma unroll
                for (int ks = 0; ks < 3; ++ks)
                    afr[ct][ks] =
                        *(const v2f*)&w1pair[ks][cb * 32 + ct * 16 + mrow][2 * kh];
            // bias for this lane's 8 channel rows, per ch-tile
            v4f bia[2][2];
            #pragma unroll
            for (int ct = 0; ct < 2; ++ct) {
                bia[ct][0] = *(const v4f*)&b1s[cb * 32 + ct * 16 + 8 * kh];
                bia[ct][1] = *(const v4f*)&b1s[cb * 32 + ct * 16 + 8 * kh + 4];
            }
            for (int pt = wv; pt < 21; pt += 8) {          // pixel tiles
                const int p = pt * 16 + mrow;              // this lane's pixel (D col)
                v2f bfr[3];
                #pragma unroll
                for (int ks = 0; ks < 3; ++ks)
                    bfr[ks] = *(const v2f*)&a_lds[p][ks * 4 + 2 * kh];
                const float keepv = keep_lds[p];
                #pragma unroll
                for (int ct = 0; ct < 2; ++ct) {
                    v8f d = {0,0,0,0,0,0,0,0};
                    d = WMMA_F32K4(afr[ct][0], bfr[0], d);
                    d = WMMA_F32K4(afr[ct][1], bfr[1], d);
                    d = WMMA_F32K4(afr[ct][2], bfr[2], d);
                    float v0 = (d[0] + bia[ct][0][0]) * keepv;
                    float v1 = (d[1] + bia[ct][0][1]) * keepv;
                    float v2 = (d[2] + bia[ct][0][2]) * keepv;
                    float v3 = (d[3] + bia[ct][0][3]) * keepv;
                    float v4 = (d[4] + bia[ct][1][0]) * keepv;
                    float v5 = (d[5] + bia[ct][1][1]) * keepv;
                    float v6 = (d[6] + bia[ct][1][2]) * keepv;
                    float v7 = (d[7] + bia[ct][1][3]) * keepv;
                    v4u pk = {pack_bf(v0, v1), pack_bf(v2, v3),
                              pack_bf(v4, v5), pack_bf(v6, v7)};
                    // lane holds ch rows 8*kh..8*kh+7 of this ch-tile
                    *(v4u*)&ys[p][ct * 16 + 8 * kh] = pk;
                }
            }
        }
        wait_g2l();
        __syncthreads();

        // (c) stage-2 WMMA: accumulate 9 taps x 32 K for this slice ----------
        #pragma unroll
        for (int tap = 0; tap < 9; ++tap) {
            const int i = tap / 3;
            const int j = tap - 3 * i;
            v16bf bf0 = load_frag(&w2s[tap][mrow][0], kh);       // co 0..15
            v16bf bf1 = load_frag(&w2s[tap][16 + mrow][0], kh);  // co 16..31
            {
                int r = wv * 2;
                v16bf a = load_frag(&ys[(r - i + 2) * 18 + (mrow - j + 2)][0], kh);
                acc00 = WMMA_BF16(a, bf0, acc00);
                acc01 = WMMA_BF16(a, bf1, acc01);
            }
            {
                int r = wv * 2 + 1;
                v16bf a = load_frag(&ys[(r - i + 2) * 18 + (mrow - j + 2)][0], kh);
                acc10 = WMMA_BF16(a, bf0, acc10);
                acc11 = WMMA_BF16(a, bf1, acc11);
            }
        }
        __syncthreads();
    }

    // ---- epilogue: (+b2) * m0, NCHW stores, 8 consecutive pixels per lane --
    const size_t outBase = (size_t)bimg * 32 * 64 * 64;
    #pragma unroll
    for (int rs = 0; rs < 2; ++rs) {
        const int r = wv * 2 + rs;
        const int P = P0 + r;
        float mk[8];
        #pragma unroll
        for (int e = 0; e < 8; ++e)
            mk[e] = (xs[r + 2][kh * 8 + e + 2] != 0.0f) ? 1.0f : 0.0f;
        #pragma unroll
        for (int nt = 0; nt < 2; ++nt) {
            const int   co   = nt * 16 + mrow;
            const float bias = b2s[co];
            const v8f acc = (rs == 0) ? ((nt == 0) ? acc00 : acc01)
                                      : ((nt == 0) ? acc10 : acc11);
            v4f s0, s1;
            #pragma unroll
            for (int e = 0; e < 4; ++e) s0[e] = (acc[e]     + bias) * mk[e];
            #pragma unroll
            for (int e = 0; e < 4; ++e) s1[e] = (acc[4 + e] + bias) * mk[4 + e];
            float* orow = outg + outBase +
                          (((size_t)co * 64 + P) * 64 + Q0 + kh * 8);
            *(v4f*)orow       = s0;
            *(v4f*)(orow + 4) = s1;
        }
    }
}

extern "C" void kernel_launch(void* const* d_in, const int* in_sizes, int n_in,
                              void* d_out, int out_size, void* d_ws, size_t ws_size,
                              hipStream_t stream) {
    const float* x  = (const float*)d_in[0];
    const float* W1 = (const float*)d_in[1];
    const float* b1 = (const float*)d_in[2];
    const float* W2 = (const float*)d_in[3];
    const float* b2 = (const float*)d_in[4];
    float* out = (float*)d_out;

    const bool use_ws = (d_ws != nullptr) && (ws_size >= 18432u * sizeof(unsigned short));
    unsigned short* w2b = use_ws ? (unsigned short*)d_ws : nullptr;
    if (use_ws) {
        prep_w2_bf16<<<72, 256, 0, stream>>>(W2, w2b);   // 18432 elements
    }

    const int B = in_sizes[0] / (64 * 64);   // NHWC with C=1
    dim3 grid(4, 4, (unsigned)B);            // 16 spatial tiles per image
    sparse_conv_fused<<<grid, 256, 0, stream>>>(x, W1, b1, W2, b2, w2b, out);

    (void)n_in; (void)out_size; (void)ws_size;
}
